// SelfAttentionLayer_5514738008259
// MI455X (gfx1250) — matrix operand order repaired and checked by reading, not compiled, not run on hardware
//
#include <hip/hip_runtime.h>
#include <hip/hip_bf16.h>
#include <stdint.h>

// Problem dims (fixed by the reference)
#define BQ   8
#define SEQ  2048
#define DIM  512
#define HDIM 512
#define MTOT (BQ * SEQ)   // 16384

// ---------------------------------------------------------------------------
// CDNA5 WMMA types
// ---------------------------------------------------------------------------
typedef __attribute__((ext_vector_type(16))) __bf16 bf16x16;
typedef __attribute__((ext_vector_type(8)))  float  f32x8;

__device__ __forceinline__ f32x8 zero8() {
  f32x8 z = {0.f, 0.f, 0.f, 0.f, 0.f, 0.f, 0.f, 0.f};
  return z;
}

// fp32 -> bf16 round-to-nearest-even, as raw bits
__device__ __forceinline__ uint16_t f2bf_bits(float f) {
  union { float f; uint32_t u; } x; x.f = f;
  uint32_t u = x.u;
  uint32_t r = u + 0x7FFFu + ((u >> 16) & 1u);
  if ((u & 0x7F800000u) == 0x7F800000u) r = u;   // inf/nan passthrough
  return (uint16_t)(r >> 16);
}

// 16 bf16 per lane = 8 VGPRs = A or B operand of v_wmma_f32_16x16x32_bf16
union FragAB {
  bf16x16 v;
  uint4   q[2];
};

// ---------------------------------------------------------------------------
// Per-wave 64(M) x 32(N) output tile (MT=4, NT=2), K-contraction in steps
// of 32, k-loop unrolled 2x (16 WMMAs / 24 b128 loads per iteration).
// Single per-lane base pointer for A and for B; every fragment row/chunk is
// a compile-time byte offset (fits signed 24-bit IOFFSET) -> only 2 live
// pointer pairs, no address spills.
//
// A fragment layout: lane L holds row m = m0 + L%16, hi = L/16;
//   element j -> k = kb + 16*(j>>3) + 8*hi + (j&7)  => chunks at +0 and +32 B.
// B fragment layout (source Wt[N][K] row-major): lane holds n = n0 + L%16;
//   element j -> k = kb + 16*(j>>3) + 8*hi + (j&7)  => chunks at +0 and +32 B.
// C/D layout: VGPR r, lane L -> m = r + 8*(L/16), n = L%16.
//
// Block = 256 threads = 8 waves arranged 2(M) x 4(N): block tile = 128x128.
// ---------------------------------------------------------------------------
template <int MT, int NT, int LDA, int LDB>
__device__ __forceinline__ void gemm_wave(const uint16_t* __restrict__ A,
                                          const uint16_t* __restrict__ Bt,
                                          int m0, int n0, int K, int lane,
                                          f32x8 (&acc)[MT][NT]) {
  const int ml = lane & 15;
  const int hi = lane >> 4;
  const uint16_t* pa = A  + (size_t)(m0 + ml) * LDA + 8 * hi;
  const uint16_t* pb = Bt + (size_t)(n0 + ml) * LDB + 8 * hi;
#pragma unroll 2
  for (int kb = 0; kb < K; kb += 32) {
    // Speculative DEV-scope prefetch one k-step ahead (translation failures
    // silently dropped per ISA) -> no branch in the loop.
    __builtin_prefetch(pa + kb + 32, 0, 1);
    __builtin_prefetch(pb + kb + 32, 0, 1);
    FragAB a[MT];
    FragAB b[NT];
#pragma unroll
    for (int i = 0; i < MT; ++i) {
      a[i].q[0] = *reinterpret_cast<const uint4*>(pa + i * (16 * LDA) + kb);
      a[i].q[1] = *reinterpret_cast<const uint4*>(pa + i * (16 * LDA) + kb + 16);
    }
#pragma unroll
    for (int j = 0; j < NT; ++j) {
      b[j].q[0] = *reinterpret_cast<const uint4*>(pb + j * (16 * LDB) + kb);
      b[j].q[1] = *reinterpret_cast<const uint4*>(pb + j * (16 * LDB) + kb + 16);
    }
#pragma unroll
    for (int j = 0; j < NT; ++j)
#pragma unroll
      for (int i = 0; i < MT; ++i)
        acc[i][j] = __builtin_amdgcn_wmma_f32_16x16x32_bf16(
            false, a[i].v, false, b[j].v, (short)0, acc[i][j], false, false);
  }
}

// ---------------------------------------------------------------------------
// K0a: x fp32 -> bf16
// ---------------------------------------------------------------------------
__global__ void k_pack_x(const float* __restrict__ x, uint16_t* __restrict__ xb,
                         int n) {
  int i = blockIdx.x * 256 + threadIdx.x;
  if (i < n) xb[i] = f2bf_bits(x[i]);
}

// ---------------------------------------------------------------------------
// K0b: pack weights. WqkvT[(3*512)][512] (transposed, concat), WoT[512][512],
//      bqkv[1536] fp32 concat.
// ---------------------------------------------------------------------------
__global__ void k_pack_w(const float* __restrict__ Wq, const float* __restrict__ Wk,
                         const float* __restrict__ Wv, const float* __restrict__ Wo,
                         const float* __restrict__ bq, const float* __restrict__ bk,
                         const float* __restrict__ bv,
                         uint16_t* __restrict__ WqkvT, uint16_t* __restrict__ WoT,
                         float* __restrict__ bqkv) {
  const int WN = DIM * HDIM;  // 262144
  int i = blockIdx.x * 256 + threadIdx.x;
  if (i < 3 * WN) {
    int which = i / WN, r = i % WN;
    int n = r / DIM;   // output feature (row of Wt)
    int k = r % DIM;   // input feature
    const float* W = (which == 0) ? Wq : (which == 1) ? Wk : Wv;
    WqkvT[(size_t)(which * HDIM + n) * DIM + k] = f2bf_bits(W[(size_t)k * HDIM + n]);
  } else if (i < 4 * WN) {
    int r = i - 3 * WN;
    int n = r / HDIM;  // d
    int k = r % HDIM;  // h
    WoT[(size_t)n * HDIM + k] = f2bf_bits(Wo[(size_t)k * DIM + n]);
  } else if (i < 4 * WN + 3 * HDIM) {
    int r = i - 4 * WN;
    int which = r / HDIM, j = r % HDIM;
    const float* bb = (which == 0) ? bq : (which == 1) ? bk : bv;
    bqkv[r] = bb[j];
  }
}

// ---------------------------------------------------------------------------
// K1: fused QKV projection. M=16384, N=1536, K=512. Wave tile 64x32.
//     Q,K stored row-major bf16 [B*S][512]; V stored transposed Vt[b][h][s].
// ---------------------------------------------------------------------------
__global__ void k_qkv(const uint16_t* __restrict__ xb,
                      const uint16_t* __restrict__ WqkvT,
                      const float* __restrict__ bqkv,
                      uint16_t* __restrict__ Qb, uint16_t* __restrict__ Kb,
                      uint16_t* __restrict__ Vt) {
  int lane = threadIdx.x & 31, w = threadIdx.x >> 5;
  int m0 = blockIdx.y * 128 + (w >> 2) * 64;
  int n0 = blockIdx.x * 128 + (w & 3) * 32;
  f32x8 acc[4][2];
#pragma unroll
  for (int i = 0; i < 4; ++i)
#pragma unroll
    for (int j = 0; j < 2; ++j) acc[i][j] = zero8();
  gemm_wave<4, 2, DIM, DIM>(xb, WqkvT, m0, n0, DIM, lane, acc);

  int hi = lane >> 4, nl = lane & 15;
  for (int mt = 0; mt < 4; ++mt)
    for (int nt = 0; nt < 2; ++nt)
      for (int r = 0; r < 8; ++r) {
        int m = m0 + mt * 16 + r + 8 * hi;
        int n = n0 + nt * 16 + nl;
        uint16_t h = f2bf_bits(acc[mt][nt][r] + bqkv[n]);
        if (n < 512) {
          Qb[(size_t)m * 512 + n] = h;
        } else if (n < 1024) {
          Kb[(size_t)m * 512 + (n - 512)] = h;
        } else {
          int b = m >> 11, s = m & 2047, hh = n - 1024;
          Vt[((size_t)b * 512 + hh) * SEQ + s] = h;
        }
      }
}

// ---------------------------------------------------------------------------
// K2: scores = Q @ K^T / sqrt(H), per batch.  K row-major == Wt[N=key][K=h].
//     Wave tile 64x32.
// ---------------------------------------------------------------------------
__global__ void k_scores(const uint16_t* __restrict__ Qb,
                         const uint16_t* __restrict__ Kb,
                         float* __restrict__ Pf) {
  int lane = threadIdx.x & 31, w = threadIdx.x >> 5;
  int b = blockIdx.z;
  const uint16_t* A  = Qb + (size_t)b * SEQ * HDIM;
  const uint16_t* Bt = Kb + (size_t)b * SEQ * HDIM;
  float* outp = Pf + (size_t)b * SEQ * SEQ;
  int m0 = blockIdx.y * 128 + (w >> 2) * 64;
  int n0 = blockIdx.x * 128 + (w & 3) * 32;
  f32x8 acc[4][2];
#pragma unroll
  for (int i = 0; i < 4; ++i)
#pragma unroll
    for (int j = 0; j < 2; ++j) acc[i][j] = zero8();
  gemm_wave<4, 2, HDIM, HDIM>(A, Bt, m0, n0, HDIM, lane, acc);

  const float scale = 0.04419417382415922f;  // 1/sqrt(512)
  int hi = lane >> 4, nl = lane & 15;
  for (int mt = 0; mt < 4; ++mt)
    for (int nt = 0; nt < 2; ++nt)
      for (int r = 0; r < 8; ++r) {
        int m = m0 + mt * 16 + r + 8 * hi;
        int n = n0 + nt * 16 + nl;
        outp[(size_t)m * SEQ + n] = acc[mt][nt][r] * scale;
      }
}

// ---------------------------------------------------------------------------
// K3: row-wise softmax.  One 256-thread block per row (2048 cols, 8/thread).
// ---------------------------------------------------------------------------
__global__ void k_softmax(const float* __restrict__ Pf, uint16_t* __restrict__ Pb) {
  size_t row = blockIdx.x;
  const float* src = Pf + row * SEQ;
  uint16_t* dst = Pb + row * SEQ;
  int t = threadIdx.x, w = t >> 5, lane = t & 31;

  float vals[8];
  float mx = -3.402823466e38f;
  for (int i = 0; i < 8; ++i) {
    vals[i] = src[t + i * 256];
    mx = fmaxf(mx, vals[i]);
  }
  for (int off = 16; off >= 1; off >>= 1) mx = fmaxf(mx, __shfl_xor(mx, off, 32));
  __shared__ float redmax[8];
  __shared__ float redsum[8];
  if (lane == 0) redmax[w] = mx;
  __syncthreads();
  mx = redmax[0];
  for (int i = 1; i < 8; ++i) mx = fmaxf(mx, redmax[i]);

  float s = 0.f;
  for (int i = 0; i < 8; ++i) {
    vals[i] = __expf(vals[i] - mx);
    s += vals[i];
  }
  for (int off = 16; off >= 1; off >>= 1) s += __shfl_xor(s, off, 32);
  if (lane == 0) redsum[w] = s;
  __syncthreads();
  s = 0.f;
  for (int i = 0; i < 8; ++i) s += redsum[i];
  float inv = 1.0f / s;
  for (int i = 0; i < 8; ++i) dst[t + i * 256] = f2bf_bits(vals[i] * inv);
}

// ---------------------------------------------------------------------------
// K4: context = P @ V, per batch.  Vt[b][h][s] == Wt[N=h][K=key].
//     Wave tile 64x32.
// ---------------------------------------------------------------------------
__global__ void k_ctx(const uint16_t* __restrict__ Pb,
                      const uint16_t* __restrict__ Vt,
                      uint16_t* __restrict__ Ctxb) {
  int lane = threadIdx.x & 31, w = threadIdx.x >> 5;
  int b = blockIdx.z;
  const uint16_t* A  = Pb + (size_t)b * SEQ * SEQ;
  const uint16_t* Bt = Vt + (size_t)b * HDIM * SEQ;
  int m0 = blockIdx.y * 128 + (w >> 2) * 64;
  int n0 = blockIdx.x * 128 + (w & 3) * 32;
  f32x8 acc[4][2];
#pragma unroll
  for (int i = 0; i < 4; ++i)
#pragma unroll
    for (int j = 0; j < 2; ++j) acc[i][j] = zero8();
  gemm_wave<4, 2, SEQ, SEQ>(A, Bt, m0, n0, SEQ, lane, acc);

  int hi = lane >> 4, nl = lane & 15;
  for (int mt = 0; mt < 4; ++mt)
    for (int nt = 0; nt < 2; ++nt)
      for (int r = 0; r < 8; ++r) {
        int m = m0 + mt * 16 + r + 8 * hi;
        int n = n0 + nt * 16 + nl;
        Ctxb[((size_t)b * SEQ + m) * HDIM + n] = f2bf_bits(acc[mt][nt][r]);
      }
}

// ---------------------------------------------------------------------------
// K5: out = context @ Wo + bo + x  (fp32 output, residual add). Wave tile 64x32.
// ---------------------------------------------------------------------------
__global__ void k_out(const uint16_t* __restrict__ Ctxb,
                      const uint16_t* __restrict__ WoT,
                      const float* __restrict__ bo,
                      const float* __restrict__ x,
                      float* __restrict__ outp) {
  int lane = threadIdx.x & 31, w = threadIdx.x >> 5;
  int m0 = blockIdx.y * 128 + (w >> 2) * 64;
  int n0 = blockIdx.x * 128 + (w & 3) * 32;
  f32x8 acc[4][2];
#pragma unroll
  for (int i = 0; i < 4; ++i)
#pragma unroll
    for (int j = 0; j < 2; ++j) acc[i][j] = zero8();
  gemm_wave<4, 2, HDIM, HDIM>(Ctxb, WoT, m0, n0, HDIM, lane, acc);

  int hi = lane >> 4, nl = lane & 15;
  for (int mt = 0; mt < 4; ++mt)
    for (int nt = 0; nt < 2; ++nt)
      for (int r = 0; r < 8; ++r) {
        int m = m0 + mt * 16 + r + 8 * hi;
        int n = n0 + nt * 16 + nl;
        size_t idx = (size_t)m * DIM + n;
        outp[idx] = acc[mt][nt][r] + bo[n] + x[idx];
      }
}

// ---------------------------------------------------------------------------
// Launch.  Workspace budget ~288 MB.
// ---------------------------------------------------------------------------
extern "C" void kernel_launch(void* const* d_in, const int* in_sizes, int n_in,
                              void* d_out, int out_size, void* d_ws, size_t ws_size,
                              hipStream_t stream) {
  (void)in_sizes; (void)n_in; (void)out_size; (void)ws_size;
  const float* x  = (const float*)d_in[0];
  const float* Wq = (const float*)d_in[1];
  const float* bq = (const float*)d_in[2];
  const float* Wk = (const float*)d_in[3];
  const float* bk = (const float*)d_in[4];
  const float* Wv = (const float*)d_in[5];
  const float* bv = (const float*)d_in[6];
  const float* Wo = (const float*)d_in[7];
  const float* bo = (const float*)d_in[8];
  float* outp = (float*)d_out;

  char* ws = (char*)d_ws;
  size_t off = 0;
  auto alloc = [&](size_t bytes) -> char* {
    char* p = ws + off;
    off = (off + bytes + 255) & ~(size_t)255;
    return p;
  };

  uint16_t* xb    = (uint16_t*)alloc((size_t)MTOT * DIM * 2);          // 16.8 MB
  uint16_t* WqkvT = (uint16_t*)alloc((size_t)3 * HDIM * DIM * 2);      //  1.6 MB
  uint16_t* WoT   = (uint16_t*)alloc((size_t)DIM * HDIM * 2);          //  0.5 MB
  float*    bqkv  = (float*)   alloc((size_t)3 * HDIM * 4);            //  6 KB
  uint16_t* Qb    = (uint16_t*)alloc((size_t)MTOT * HDIM * 2);         // 16.8 MB
  uint16_t* Kb    = (uint16_t*)alloc((size_t)MTOT * HDIM * 2);         // 16.8 MB
  uint16_t* Vt    = (uint16_t*)alloc((size_t)MTOT * HDIM * 2);         // 16.8 MB
  float*    Pf    = (float*)   alloc((size_t)BQ * SEQ * SEQ * 4);      // 134 MB
  uint16_t* Pb    = (uint16_t*)alloc((size_t)BQ * SEQ * SEQ * 2);      //  67 MB
  uint16_t* Ctxb  = (uint16_t*)alloc((size_t)MTOT * HDIM * 2);         // 16.8 MB

  // pack
  {
    int n = MTOT * DIM;
    k_pack_x<<<(n + 255) / 256, 256, 0, stream>>>(x, xb, n);
    int nw = 4 * DIM * HDIM + 3 * HDIM;
    k_pack_w<<<(nw + 255) / 256, 256, 0, stream>>>(Wq, Wk, Wv, Wo, bq, bk, bv,
                                                   WqkvT, WoT, bqkv);
  }
  // QKV projection: M=16384, N=1536, block tile 128x128
  k_qkv<<<dim3(1536 / 128, MTOT / 128, 1), 256, 0, stream>>>(xb, WqkvT, bqkv,
                                                             Qb, Kb, Vt);
  // scores: per batch, M=N=2048, block tile 128x128
  k_scores<<<dim3(SEQ / 128, SEQ / 128, BQ), 256, 0, stream>>>(Qb, Kb, Pf);
  // softmax: one block per row
  k_softmax<<<dim3(MTOT, 1, 1), 256, 0, stream>>>(Pf, Pb);
  // context: per batch, M=2048, N=512, block tile 128x128
  k_ctx<<<dim3(HDIM / 128, SEQ / 128, BQ), 256, 0, stream>>>(Pb, Vt, Ctxb);
  // output projection + residual: M=16384, N=512, block tile 128x128
  k_out<<<dim3(DIM / 128, MTOT / 128, 1), 256, 0, stream>>>(Ctxb, WoT, bo, x, outp);
}
